// MoeLayer_1752346657110
// MI455X (gfx1250) — compile-verified, matrix operand order checked by reading, not run on hardware
//
#include <hip/hip_runtime.h>
#include <hip/hip_bf16.h>

typedef __attribute__((ext_vector_type(16))) __bf16 v16bf;
typedef __attribute__((ext_vector_type(8)))  float  v8f;

#define T_TOK  8192
#define DMODEL 2048
#define HDIM   7168
#define NEXP   8

// ---------------------------------------------------------------------------
// CDNA5 async global->LDS staging (ASYNCcnt path, CDNA5 ISA)
// vdst = 32-bit LDS byte address, vaddr = 64-bit global address
// ---------------------------------------------------------------------------
__device__ __forceinline__ void async_load_b128(unsigned lds_addr,
                                                const void* gptr) {
  unsigned long long ga = (unsigned long long)(size_t)gptr;
  asm volatile("global_load_async_to_lds_b128 %0, %1, off"
               :: "v"(lds_addr), "v"(ga) : "memory");
}

__device__ __forceinline__ void wait_async0() {
#if __has_builtin(__builtin_amdgcn_s_wait_asynccnt)
  __builtin_amdgcn_s_wait_asynccnt(0);
#else
  asm volatile("s_wait_asynccnt 0x0" ::: "memory");
#endif
}

__device__ __forceinline__ unsigned lds_addr_of(const void* p) {
  return (unsigned)(size_t)p;   // generic LDS pointer: low 32 bits = LDS offset
}

// ---------------------------------------------------------------------------
// WMMA bf16 fragment builders (16-bit operand VGPR layout, wave32)
// A 16x32: lane L -> M = L&15, half = L>>4; vgpr-slot v holds the K pair:
//   v<4 : K = half*8 + 2v   ;  v>=4 : K = 16 + half*8 + 2(v-4)
// B 32x16 mirrors with N = L&15.
// ---------------------------------------------------------------------------
__device__ __forceinline__ int koff_v(int v, int half) {
  return (v < 4) ? (half * 8 + v * 2) : (16 + half * 8 + (v - 4) * 2);
}

// row-major fp32 source (global), consecutive K: convert pairs to bf16
__device__ __forceinline__ v16bf load_frag_f32row(const float* __restrict__ row,
                                                  int k0, int half) {
  v16bf f;
#pragma unroll
  for (int v = 0; v < 8; ++v) {
    int ko = koff_v(v, half);
    float2 p = *(const float2*)(row + k0 + ko);
    f[2 * v]     = (__bf16)p.x;
    f[2 * v + 1] = (__bf16)p.y;
  }
  return f;
}

// row-major bf16 source (global h buffer), consecutive K
__device__ __forceinline__ v16bf load_frag_bf16row(const __bf16* __restrict__ row,
                                                   int k0, int half) {
  v16bf f;
#pragma unroll
  for (int v = 0; v < 8; ++v) {
    int ko = koff_v(v, half);
    f[2 * v]     = row[k0 + ko];
    f[2 * v + 1] = row[k0 + ko + 1];
  }
  return f;
}

// LDS fp32 row (consecutive K): ds_load_b64 pairs -> bf16
__device__ __forceinline__ v16bf frag_from_lds_row(const float* lrow, int half) {
  v16bf f;
#pragma unroll
  for (int v = 0; v < 8; ++v) {
    int ko = koff_v(v, half);
    float2 p = *(const float2*)(lrow + ko);
    f[2 * v]     = (__bf16)p.x;
    f[2 * v + 1] = (__bf16)p.y;
  }
  return f;
}

// LDS fp32 tile walked down a column (fixed col, K strided by `stride`)
__device__ __forceinline__ v16bf frag_from_lds_col(const float* base, int half,
                                                   int stride, int col) {
  v16bf f;
#pragma unroll
  for (int v = 0; v < 8; ++v) {
    int ko = koff_v(v, half);
    f[2 * v]     = (__bf16)base[ko * stride + col];
    f[2 * v + 1] = (__bf16)base[(ko + 1) * stride + col];
  }
  return f;
}

// ---------------------------------------------------------------------------
// zero-init (out buffer, counters)
// ---------------------------------------------------------------------------
__global__ void zero_kernel(float* __restrict__ p, int n) {
  int i = blockIdx.x * blockDim.x + threadIdx.x;
  if (i < n) p[i] = 0.0f;
}

// ---------------------------------------------------------------------------
// gating: logits = x @ gate_w^T (E=8), top-2, softmax, scatter to expert lists
// one wave (32 lanes) per token
// ---------------------------------------------------------------------------
__global__ void __launch_bounds__(256)
gate_kernel(const float* __restrict__ x, const float* __restrict__ gate_w,
            int* __restrict__ cnt, int* __restrict__ rowlist,
            float* __restrict__ wgtlist) {
  int wave = threadIdx.x >> 5;
  int lane = threadIdx.x & 31;
  int t = blockIdx.x * 8 + wave;
  if (t >= T_TOK) return;

  const float* xr = x + (size_t)t * DMODEL;
  float acc[NEXP];
#pragma unroll
  for (int e = 0; e < NEXP; ++e) acc[e] = 0.0f;

  for (int k = lane; k < DMODEL; k += 32) {
    float xv = xr[k];
#pragma unroll
    for (int e = 0; e < NEXP; ++e)
      acc[e] = fmaf(xv, gate_w[e * DMODEL + k], acc[e]);
  }
#pragma unroll
  for (int e = 0; e < NEXP; ++e) {
#pragma unroll
    for (int off = 16; off > 0; off >>= 1)
      acc[e] += __shfl_xor(acc[e], off, 32);
  }

  if (lane == 0) {
    int i0 = 0; float v0 = acc[0];
#pragma unroll
    for (int e = 1; e < NEXP; ++e)
      if (acc[e] > v0) { v0 = acc[e]; i0 = e; }
    int i1 = -1; float v1 = -3.4e38f;
#pragma unroll
    for (int e = 0; e < NEXP; ++e)
      if (e != i0 && acc[e] > v1) { v1 = acc[e]; i1 = e; }

    float e1 = __expf(v1 - v0);          // softmax over {v0, v1}
    float w0 = __builtin_amdgcn_rcpf(1.0f + e1);
    float w1 = e1 * w0;

    int p0 = atomicAdd(&cnt[i0], 1);
    rowlist[i0 * T_TOK + p0] = t * 2 + 0;
    wgtlist[i0 * T_TOK + p0] = w0;
    int p1 = atomicAdd(&cnt[i1], 1);
    rowlist[i1 * T_TOK + p1] = t * 2 + 1;
    wgtlist[i1 * T_TOK + p1] = w1;
  }
}

// ---------------------------------------------------------------------------
// phase 1: h[row] = silu(x @ w1[e]^T) * (x @ w3[e]^T)  (bf16 result)
// block = 8 waves, tile 32 tokens x 64 H cols, K loop over D in steps of 32.
// w1/w3 K-slices (64 rows x 32 K fp32 each) are double-buffer staged into LDS
// with global_load_async_to_lds_b128 (ASYNCcnt), A (gathered x rows) streams
// straight from global. Each wave runs a 16x16 WMMA tile with dual (c1,c3)
// accumulators sharing the A fragment.
// ---------------------------------------------------------------------------
__global__ void __launch_bounds__(256)
moe_h_kernel(const float* __restrict__ x, const float* __restrict__ w1,
             const float* __restrict__ w3, const int* __restrict__ cnt,
             const int* __restrict__ rowlist, __bf16* __restrict__ hbuf) {
  int e = blockIdx.y;
  int n = cnt[e];
  int mtile = blockIdx.x;
  if (mtile * 32 >= n) return;
  int ntile = blockIdx.z;
  int tid = threadIdx.x;

  __shared__ int   s_row[32];
  __shared__ float sB[2][2][64][32];     // [buf][w1|w3][row][k], 32 KB

  if (tid < 32) {
    int slot = mtile * 32 + tid;
    if (slot >= n) slot = n - 1;                 // duplicate last valid row
    s_row[tid] = rowlist[e * T_TOK + slot];
  }
  __syncthreads();

  int wave = tid >> 5;
  int lane = tid & 31;
  int wm = wave & 1, wn = wave >> 1;
  int half = lane >> 4, mi = lane & 15;

  const float* arow = x + (size_t)(s_row[wm * 16 + mi] >> 1) * DMODEL;
  size_t wbase = (size_t)e * HDIM * DMODEL;
  const float* b1t = w1 + wbase + (size_t)(ntile * 64) * DMODEL; // tile row 0
  const float* b3t = w3 + wbase + (size_t)(ntile * 64) * DMODEL;

  unsigned sB0 = lds_addr_of(&sB[0][0][0][0]);
  const unsigned BUFBYTES = 2u * 64u * 32u * 4u;   // one buffer (both mats)
  const unsigned MATBYTES = 64u * 32u * 4u;

  // per-thread staging assignment: 512 x b128 per matrix per step
  int id0 = tid * 2, id1 = tid * 2 + 1;

#define STAGE_W13(bufsel, kk)                                                  \
  {                                                                            \
    unsigned base = sB0 + (bufsel) * BUFBYTES;                                 \
    int ids[2] = {id0, id1};                                                   \
    _Pragma("unroll") for (int j = 0; j < 2; ++j) {                            \
      int row = ids[j] >> 3, chunk = ids[j] & 7;                               \
      unsigned loff = (unsigned)(row * 32 + chunk * 4) * 4u;                   \
      async_load_b128(base + loff,                                             \
                      b1t + (size_t)row * DMODEL + (kk) + chunk * 4);          \
      async_load_b128(base + MATBYTES + loff,                                  \
                      b3t + (size_t)row * DMODEL + (kk) + chunk * 4);          \
    }                                                                          \
  }

  STAGE_W13(0, 0);                        // prologue fill of buffer 0

  v8f c1 = {}, c3 = {};
  const int NK = DMODEL / 32;
  for (int kt = 0; kt < NK; ++kt) {
    int k0 = kt * 32;
    int buf = kt & 1;
    wait_async0();                        // own fill(kt) landed
    __syncthreads();                      // barrier A: buffer kt ready
    v16bf a  = load_frag_f32row(arow, k0, half);
    v16bf b1 = frag_from_lds_row(&sB[buf][0][wn * 16 + mi][0], half);
    v16bf b3 = frag_from_lds_row(&sB[buf][1][wn * 16 + mi][0], half);
    __syncthreads();                      // barrier B: frees buf^1 for refill
    if (kt + 1 < NK) {
      STAGE_W13(buf ^ 1, k0 + 32);        // overlap next stage with WMMA
      __builtin_prefetch(arow + k0 + 32, 0, 3);   // global_prefetch_b8
    }
    c1 = __builtin_amdgcn_wmma_f32_16x16x32_bf16(false, a, false, b1,
                                                 (short)0, c1, false, false);
    c3 = __builtin_amdgcn_wmma_f32_16x16x32_bf16(false, a, false, b3,
                                                 (short)0, c3, false, false);
  }
#undef STAGE_W13

  int col = ntile * 64 + wn * 16 + mi;    // C/D: N = lane&15
#pragma unroll
  for (int r = 0; r < 8; ++r) {           // C/D: M = r + 8*half
    int mloc = wm * 16 + r + 8 * half;
    int slot = mtile * 32 + mloc;
    if (slot < n) {
      float a1 = c1[r];
      float s  = a1 * __builtin_amdgcn_rcpf(1.0f + __expf(-a1));  // silu
      float hv = s * c3[r];
      hbuf[(size_t)s_row[mloc] * HDIM + col] = (__bf16)hv;
    }
  }
}

// ---------------------------------------------------------------------------
// phase 2: out[token] += gate_weight * (h[row] @ w2[e]),  w2[e] is [H, D]
// tile 32 rows x 64 D cols, K loop over H in steps of 32. The w2 K-slice
// (32 K rows x 64 D cols fp32) is async-staged into LDS (coalesced row-wise
// B128s) so B fragments become cheap LDS column walks instead of stride-D
// global scalar loads. fp32 atomics into out.
// ---------------------------------------------------------------------------
__global__ void __launch_bounds__(256)
moe_out_kernel(const __bf16* __restrict__ hbuf, const float* __restrict__ w2,
               const int* __restrict__ cnt, const int* __restrict__ rowlist,
               const float* __restrict__ wgtlist, float* __restrict__ out) {
  int e = blockIdx.y;
  int n = cnt[e];
  int mtile = blockIdx.x;
  if (mtile * 32 >= n) return;
  int dtile = blockIdx.z;
  int tid = threadIdx.x;

  __shared__ int   s_row[32];
  __shared__ float s_w[32];
  __shared__ float sW2[2][32][64];        // [buf][k][dcol], 16 KB

  if (tid < 32) {
    int slot = mtile * 32 + tid;
    int cl = slot >= n ? n - 1 : slot;
    s_row[tid] = rowlist[e * T_TOK + cl];
    s_w[tid]   = wgtlist[e * T_TOK + cl];
  }
  __syncthreads();

  int wave = tid >> 5;
  int lane = tid & 31;
  int wm = wave & 1, wn = wave >> 1;
  int half = lane >> 4, mi = lane & 15;

  const __bf16* arow = hbuf + (size_t)s_row[wm * 16 + mi] * HDIM;
  int dcol0 = dtile * 64;
  const float* w2t = w2 + (size_t)e * HDIM * DMODEL + dcol0;  // [H, 64-slice]

  unsigned sW0 = lds_addr_of(&sW2[0][0][0]);
  const unsigned BUFBYTES2 = 32u * 64u * 4u;
  int id0 = tid * 2, id1 = tid * 2 + 1;   // 512 x b128 per step

#define STAGE_W2(bufsel, kk)                                                   \
  {                                                                            \
    unsigned base = sW0 + (bufsel) * BUFBYTES2;                                \
    int ids[2] = {id0, id1};                                                   \
    _Pragma("unroll") for (int j = 0; j < 2; ++j) {                            \
      int row = ids[j] >> 4, chunk = ids[j] & 15;                              \
      async_load_b128(base + (unsigned)(row * 64 + chunk * 4) * 4u,            \
                      w2t + (size_t)((kk) + row) * DMODEL + chunk * 4);        \
    }                                                                          \
  }

  STAGE_W2(0, 0);

  v8f c = {};
  const int NK = HDIM / 32;
  for (int kt = 0; kt < NK; ++kt) {
    int k0 = kt * 32;
    int buf = kt & 1;
    wait_async0();
    __syncthreads();                      // barrier A
    v16bf a = load_frag_bf16row(arow, k0, half);
    v16bf b = frag_from_lds_col(&sW2[buf][0][0], half, 64, wn * 16 + mi);
    __syncthreads();                      // barrier B
    if (kt + 1 < NK) {
      STAGE_W2(buf ^ 1, k0 + 32);
      __builtin_prefetch(arow + k0 + 32, 0, 3);
    }
    c = __builtin_amdgcn_wmma_f32_16x16x32_bf16(false, a, false, b,
                                                (short)0, c, false, false);
  }
#undef STAGE_W2

#pragma unroll
  for (int r = 0; r < 8; ++r) {
    int mloc = wm * 16 + r + 8 * half;
    int slot = mtile * 32 + mloc;
    if (slot < n) {
      int token = s_row[mloc] >> 1;
      float val = c[r] * s_w[mloc];
      unsafeAtomicAdd(&out[(size_t)token * DMODEL + dcol0 + wn * 16 + mi],
                      val);              // global_atomic_add_f32
    }
  }
}

// ---------------------------------------------------------------------------
// launch: zero -> gate/route -> grouped GEMM1 (h) -> grouped GEMM2 (out)
// workspace: [cnt:256B][rowlist:E*T*4][wgtlist:E*T*4][hbuf:T*2*H*2]  (~235 MB)
// ---------------------------------------------------------------------------
extern "C" void kernel_launch(void* const* d_in, const int* in_sizes, int n_in,
                              void* d_out, int out_size, void* d_ws,
                              size_t ws_size, hipStream_t stream) {
  const float* x      = (const float*)d_in[0];
  const float* gate_w = (const float*)d_in[1];
  const float* w1     = (const float*)d_in[2];
  const float* w2     = (const float*)d_in[3];
  const float* w3     = (const float*)d_in[4];
  float* out = (float*)d_out;

  char* ws = (char*)d_ws;
  int*    cnt     = (int*)ws;
  int*    rowlist = (int*)(ws + 256);
  float*  wgtlist = (float*)(ws + 256 + (size_t)NEXP * T_TOK * 4);
  __bf16* hbuf    = (__bf16*)(ws + 256 + (size_t)2 * NEXP * T_TOK * 4);

  zero_kernel<<<(T_TOK * DMODEL + 255) / 256, 256, 0, stream>>>(out,
                                                                T_TOK * DMODEL);
  zero_kernel<<<1, 64, 0, stream>>>((float*)cnt, NEXP);

  gate_kernel<<<T_TOK / 8, 256, 0, stream>>>(x, gate_w, cnt, rowlist, wgtlist);

  dim3 g1(T_TOK / 32, NEXP, HDIM / 64);
  moe_h_kernel<<<g1, 256, 0, stream>>>(x, w1, w3, cnt, rowlist, hbuf);

  dim3 g2(T_TOK / 32, NEXP, DMODEL / 64);
  moe_out_kernel<<<g2, 256, 0, stream>>>(hbuf, w2, cnt, rowlist, wgtlist, out);
}
